// GCN_E_16801912062644
// MI455X (gfx1250) — matrix operand order repaired and checked by reading, not compiled, not run on hardware
//
#include <hip/hip_runtime.h>

typedef __attribute__((ext_vector_type(16))) __bf16 v16bf;
typedef __attribute__((ext_vector_type(8)))  float  v8f;

union FragU { uint4 q[2]; v16bf v; };

// ---------------------------------------------------------------------------
// Elementwise helpers (dout is a power of two -> mask, no division)
// ---------------------------------------------------------------------------
__global__ void k_f32_to_bf16(const float* __restrict__ in, __bf16* __restrict__ out,
                              long long n) {
    long long i = (long long)blockIdx.x * blockDim.x + threadIdx.x;
    if (i < n) out[i] = (__bf16)in[i];
}

// Weight transpose + bf16 convert: Wt[n*K + k] = bf16(W[k*N + n]).
// Tiny (<=128K elems), runs once per layer; read coalesced over n.
__global__ void k_w_transpose_bf16(const float* __restrict__ W, __bf16* __restrict__ Wt,
                                   int K, int N) {
    int i = blockIdx.x * blockDim.x + threadIdx.x;   // i = k*N + n
    if (i >= K * N) return;
    int k = i / N, n = i - k * N;
    Wt[(size_t)n * K + k] = (__bf16)W[i];
}

__global__ void k_zero_f32(float* __restrict__ p, long long n) {
    long long i = (long long)blockIdx.x * blockDim.x + threadIdx.x;
    if (i < n) p[i] = 0.0f;
}

__global__ void k_bias_leaky_to_bf16(const float* __restrict__ H,
                                     const float* __restrict__ bias,
                                     __bf16* __restrict__ out,
                                     long long n, int dmask) {
    long long i = (long long)blockIdx.x * blockDim.x + threadIdx.x;
    if (i >= n) return;
    float v = H[i] + bias[(int)i & dmask];
    v = (v > 0.0f) ? v : 0.25f * v;
    out[i] = (__bf16)v;
}

__global__ void k_bias_leaky_to_f32(const float* __restrict__ H,
                                    const float* __restrict__ bias,
                                    float* __restrict__ out,
                                    long long n, int dmask) {
    long long i = (long long)blockIdx.x * blockDim.x + threadIdx.x;
    if (i >= n) return;
    float v = H[i] + bias[(int)i & dmask];
    v = (v > 0.0f) ? v : 0.25f * v;
    out[i] = v;
}

// ---------------------------------------------------------------------------
// Dense GEMM: C[M,NOUT] = A[M,K](bf16) @ Bt[NOUT,K](bf16, pre-transposed),
// f32 accumulate. Compile-time K/NOUT -> constant strides, immediate offsets.
// Block tile 64x64, 256 threads = 8 waves (4 M-slices x 2 N-slices).
// Each wave: 16x32 output = two v_wmma_f32_16x16x32_bf16 per K-step sharing
// one A fragment; every fragment is a contiguous 16B LDS chunk (ds_load_b128).
// ---------------------------------------------------------------------------
template <int K, int NOUT>
__global__ __launch_bounds__(256)
void k_gemm_bf16_wmma(const __bf16* __restrict__ A, const __bf16* __restrict__ Bt,
                      float* __restrict__ C, int M) {
    constexpr int TM = 64, TN = 64, TK = 32;
    constexpr int LDA = 40;   // 80B row stride: 16B-aligned, conflict-free rows
    constexpr int LDB = 40;

    __shared__ __bf16 As [TM * LDA];   // As[m][k]
    __shared__ __bf16 Bst[TN * LDB];   // Bst[n][k]

    const int tid  = threadIdx.x;
    const int lane = tid & 31;
    const int wave = tid >> 5;          // 0..7
    const int wm   = wave >> 1;         // 0..3 : 16-row slice
    const int wn   = wave & 1;          // 0..1 : 32-col slice
    const int half = lane >> 4;         // wave32 half
    const int l16  = lane & 15;

    const int bm = blockIdx.x * TM;
    const int bn = blockIdx.y * TN;

    // A cooperative load: 4 threads/row, 8 contiguous bf16 (16B) each
    const int arow = tid >> 2, acol = (tid & 3) * 8;
    // B cooperative load: thread owns column n = tid&63, 8 contiguous K from Bt
    const int bncol = tid & 63, bkc = (tid >> 6) * 8;

    const __bf16* asrc = A  + (size_t)(bm + arow)  * K + acol;
    const __bf16* bsrc = Bt + (size_t)(bn + bncol) * K + bkc;
    const bool arow_ok = (bm + arow) < M;

    v8f acc0 = {}, acc1 = {};

    for (int k0 = 0; k0 < K; k0 += TK) {
        // ---- stage A tile (64x32), one 16B load/store per thread ----
        {
            uint4 aval = {0u, 0u, 0u, 0u};
            if (arow_ok) {
                aval = *reinterpret_cast<const uint4*>(asrc + k0);
                if (k0 + TK < K) __builtin_prefetch(asrc + k0 + TK, 0, 0);
            }
            *reinterpret_cast<uint4*>(&As[arow * LDA + acol]) = aval;
        }
        // ---- stage B tile (64 cols x 32 K) from pre-transposed weights ----
        *reinterpret_cast<uint4*>(&Bst[bncol * LDB + bkc]) =
            *reinterpret_cast<const uint4*>(bsrc + k0);
        __syncthreads();

        // ---- A fragment: row wm*16+l16; lanes<16 K 0-7/16-23, lanes>=16 +8 ----
        FragU fa;
        {
            const __bf16* ap = &As[(wm * 16 + l16) * LDA + half * 8];
            fa.q[0] = *reinterpret_cast<const uint4*>(ap);        // K kb..kb+7
            fa.q[1] = *reinterpret_cast<const uint4*>(ap + 16);   // K kb+16..kb+23
        }
        // ---- B fragments: same K order as A (lanes<16: K0-7/16-23, lanes>=16 +8) ----
        FragU fb0, fb1;
        {
            const __bf16* bp0 = &Bst[(wn * 32 + l16) * LDB + half * 8];
            fb0.q[0] = *reinterpret_cast<const uint4*>(bp0);
            fb0.q[1] = *reinterpret_cast<const uint4*>(bp0 + 16);
            const __bf16* bp1 = &Bst[(wn * 32 + 16 + l16) * LDB + half * 8];
            fb1.q[0] = *reinterpret_cast<const uint4*>(bp1);
            fb1.q[1] = *reinterpret_cast<const uint4*>(bp1 + 16);
        }

        acc0 = __builtin_amdgcn_wmma_f32_16x16x32_bf16(
                   false, fa.v, false, fb0.v, (short)0, acc0, false, false);
        acc1 = __builtin_amdgcn_wmma_f32_16x16x32_bf16(
                   false, fa.v, false, fb1.v, (short)0, acc1, false, false);
        __syncthreads();
    }

    // ---- store C: VGPR v -> row wm*16 + half*8 + v; cols ccol, ccol+16 ----
    const int crow0 = bm + wm * 16 + half * 8;
    const int ccol  = bn + wn * 32 + l16;
    float* cp = C + (size_t)crow0 * NOUT + ccol;
    if (bm + TM <= M) {          // block-uniform fast path: no per-row guard
        #pragma unroll
        for (int v = 0; v < 8; ++v) {
            cp[(size_t)v * NOUT]      = acc0[v];
            cp[(size_t)v * NOUT + 16] = acc1[v];
        }
    } else {
        #pragma unroll
        for (int v = 0; v < 8; ++v) {
            if (crow0 + v < M) {
                cp[(size_t)v * NOUT]      = acc0[v];
                cp[(size_t)v * NOUT + 16] = acc1[v];
            }
        }
    }
}

// ---------------------------------------------------------------------------
// Edge-parallel SpMM scatter: H[rows[e], :] += vals[e] * S[cols[e], :]
// Consecutive lanes cover consecutive float4 chunks of one edge's row:
// coalesced gathers (support matrix is L2-resident) + f32 global atomics.
// cshift = log2(dout/4).
// ---------------------------------------------------------------------------
__global__ void k_spmm_scatter(const int* __restrict__ rows,
                               const int* __restrict__ cols,
                               const float* __restrict__ vals,
                               const float* __restrict__ S,
                               float* __restrict__ H,
                               long long total, int dout, int cshift) {
    long long gid = (long long)blockIdx.x * blockDim.x + threadIdx.x;
    if (gid >= total) return;
    const int e = (int)(gid >> cshift);
    const int c = (int)(gid & ((1 << cshift) - 1));
    const int r  = rows[e];
    const int cl = cols[e];
    const float v = vals[e];
    const float4 s = reinterpret_cast<const float4*>(S + (size_t)cl * dout)[c];
    float* dst = H + (size_t)r * dout + (size_t)c * 4;
    atomicAdd(dst + 0, v * s.x);
    atomicAdd(dst + 1, v * s.y);
    atomicAdd(dst + 2, v * s.z);
    atomicAdd(dst + 3, v * s.w);
}

// ---------------------------------------------------------------------------
// Host-side orchestration
// ---------------------------------------------------------------------------
extern "C" void kernel_launch(void* const* d_in, const int* in_sizes, int n_in,
                              void* d_out, int out_size, void* d_ws, size_t ws_size,
                              hipStream_t stream) {
    (void)in_sizes; (void)n_in; (void)out_size; (void)ws_size;

    const float* x    = (const float*)d_in[0];
    const int*   rows = (const int*)  d_in[1];
    const int*   cols = (const int*)  d_in[2];
    const float* vals = (const float*)d_in[3];
    const float* W1   = (const float*)d_in[4];
    const float* b1   = (const float*)d_in[5];
    const float* W2   = (const float*)d_in[6];
    const float* b2   = (const float*)d_in[7];
    const float* W3   = (const float*)d_in[8];
    const float* b3   = (const float*)d_in[9];
    float* out = (float*)d_out;

    const int M = 100000, nnz = 3200000;

    // workspace layout (bytes)
    const size_t A_BYTES = (size_t)M * 512 * 2;   // bf16 activations (max)
    const size_t S_BYTES = (size_t)M * 256 * 4;   // f32 GEMM result (max)
    const size_t H_BYTES = (size_t)M * 256 * 4;   // f32 scatter accumulator (max)
    char* ws = (char*)d_ws;
    __bf16* wsA = (__bf16*)(ws);
    float*  wsS = (float*) (ws + A_BYTES);
    float*  wsH = (float*) (ws + A_BYTES + S_BYTES);
    __bf16* wsW = (__bf16*)(ws + A_BYTES + S_BYTES + H_BYTES);   // transposed weights

    auto blocks = [](long long n) { return (unsigned)((n + 255) / 256); };

    // stage 0: x -> bf16
    {
        long long n = (long long)M * 512;
        k_f32_to_bf16<<<blocks(n), 256, 0, stream>>>(x, wsA, n);
    }

    const unsigned gx = (unsigned)((M + 63) / 64);

    for (int l = 0; l < 3; ++l) {
        const float* W = (l == 0) ? W1 : (l == 1) ? W2 : W3;
        const float* b = (l == 0) ? b1 : (l == 1) ? b2 : b3;
        const int Kin  = (l == 0) ? 512 : 256;
        const int Dout = (l == 2) ? 128 : 256;

        // weights -> transposed bf16 (Wt[n][k])
        k_w_transpose_bf16<<<blocks((long long)Kin * Dout), 256, 0, stream>>>(
            W, wsW, Kin, Dout);

        // dense GEMM (WMMA bf16), block tile 64x64, compile-time strides
        dim3 gg(gx, Dout / 64);
        if (l == 0)      k_gemm_bf16_wmma<512, 256><<<gg, 256, 0, stream>>>(wsA, wsW, wsS, M);
        else if (l == 1) k_gemm_bf16_wmma<256, 256><<<gg, 256, 0, stream>>>(wsA, wsW, wsS, M);
        else             k_gemm_bf16_wmma<256, 128><<<gg, 256, 0, stream>>>(wsA, wsW, wsS, M);

        // zero scatter accumulator
        long long hn = (long long)M * Dout;
        k_zero_f32<<<blocks(hn), 256, 0, stream>>>(wsH, hn);

        // sparse scatter (dout/4 chunks per edge; power of two)
        const int cshift = (Dout == 256) ? 6 : 5;
        long long tot = (long long)nnz << cshift;
        k_spmm_scatter<<<blocks(tot), 256, 0, stream>>>(rows, cols, vals, wsS, wsH,
                                                        tot, Dout, cshift);

        // bias + leaky relu (fused with bf16 conversion for next layer)
        const int dmask = Dout - 1;
        if (l == 2) {
            k_bias_leaky_to_f32<<<blocks(hn), 256, 0, stream>>>(wsH, b, out, hn, dmask);
        } else {
            k_bias_leaky_to_bf16<<<blocks(hn), 256, 0, stream>>>(wsH, b, wsA, hn, dmask);
        }
    }
}
